// PureCartesianE3Conv_19172734009891
// MI455X (gfx1250) — matrix operand order repaired and checked by reading, not compiled, not run on hardware
//
#include <hip/hip_runtime.h>
#include <hip/hip_bf16.h>
#include <math.h>
#include <stdint.h>

typedef __bf16 bf16_t;
typedef bf16_t v16bf __attribute__((ext_vector_type(16)));
typedef bf16_t v8bf  __attribute__((ext_vector_type(8)));
typedef float  v8f   __attribute__((ext_vector_type(8)));
typedef float  v4f   __attribute__((ext_vector_type(4)));

#define WPB 8            // waves per block (256 threads, wave32)
#define OUTW 416         // 2*(16*1 + 16*3 + 16*9)

__device__ __forceinline__ float silu_f(float x) {
    // x * 1/(1+exp(-x)) with v_rcp_f32 (no precise-div fixup chain)
    return x * __builtin_amdgcn_rcpf(1.0f + __expf(-x));
}

// ---------------- init: zero output + per-node edge counts ----------------
__global__ void zero_kernel(float* __restrict__ out, int n_out,
                            float* __restrict__ cnt, int n_cnt) {
    int i = blockIdx.x * blockDim.x + threadIdx.x;
    if (i < n_out) out[i] = 0.0f;
    if (i < n_cnt) cnt[i] = 0.0f;
}

// ---------------- node MLP: Ai = silu(emb[A]@Wa1+ba1)@Wa2+ba2 -------------
__global__ void node_mlp_kernel(const int* __restrict__ A,
                                const float* __restrict__ emb_table,
                                const float* __restrict__ Wa1,
                                const float* __restrict__ ba1,
                                const float* __restrict__ Wa2,
                                const float* __restrict__ ba2,
                                float* __restrict__ Ai, int N) {
    int n = blockIdx.x * blockDim.x + threadIdx.x;
    if (n >= N) return;
    int a = A[n];
    const float* e = emb_table + a * 16;
    float emb[16];
#pragma unroll
    for (int k = 0; k < 16; k++) emb[k] = e[k];
    float acc[8];
#pragma unroll
    for (int c = 0; c < 8; c++) acc[c] = ba2[c];
#pragma unroll 1
    for (int j = 0; j < 64; j++) {
        float s = ba1[j];
#pragma unroll
        for (int k = 0; k < 16; k++) s += emb[k] * Wa1[k * 64 + j];
        float h = silu_f(s);
#pragma unroll
        for (int c = 0; c < 8; c++) acc[c] += h * Wa2[j * 8 + c];
    }
#pragma unroll
    for (int c = 0; c < 8; c++) Ai[n * 8 + c] = acc[c];
}

// ---------------- weight pack: f32 -> bf16 in exact WMMA-B lane order ------
// B tile (32x16): lane l (hi=l>>4, m=l&15) holds elems i=0..15 with
// K = kc*32 + 8*hi + i + (i>=8 ? 8 : 0) (pw1: hi*16 + i), N = tile*16 + m.  Wf3 cut to its first 384 columns
// (only paths p1=p2=0 contribute; the rest multiply exact zeros).
__global__ void pack_weights_kernel(const float* __restrict__ Wf1,
                                    const float* __restrict__ Wf2,
                                    const float* __restrict__ Wf3,
                                    bf16_t* __restrict__ pw1,
                                    bf16_t* __restrict__ pw2,
                                    bf16_t* __restrict__ pw3) {
    int idx = blockIdx.x * blockDim.x + threadIdx.x;
    const int n1 = 4 * 512, n2 = 8 * 512, n3 = 48 * 512;
    if (idx >= n1 + n2 + n3) return;
    if (idx < n1) {
        int t = idx >> 9, r = idx & 511;
        int lane = r >> 4, i = r & 15, hi = lane >> 4, m = lane & 15;
        int K = hi * 16 + i;                 // single K-chunk, zero-padded K>=8
        int N = t * 16 + m;
        pw1[idx] = (bf16_t)((K < 8) ? Wf1[K * 64 + N] : 0.0f);
    } else if (idx < n1 + n2) {
        int rel = idx - n1;
        int t = rel >> 9, r = rel & 511;
        int lane = r >> 4, i = r & 15, hi = lane >> 4, m = lane & 15;
        int kc = t >> 2, n = t & 3;
        int K = kc * 32 + hi * 8 + i + ((i >= 8) ? 8 : 0);
        int N = n * 16 + m;
        pw2[rel] = (bf16_t)Wf2[K * 64 + N];
    } else {
        int rel = idx - n1 - n2;
        int t2 = rel >> 9, r = rel & 511;
        int lane = r >> 4, i = r & 15, hi = lane >> 4, m = lane & 15;
        int kc = t2 / 24, t = t2 % 24;
        int K = kc * 32 + hi * 8 + i + ((i >= 8) ? 8 : 0);
        int N = t * 16 + m;                  // N < 384 of 1536
        pw3[rel] = (bf16_t)Wf3[K * 1536 + N];
    }
}

// layer-3 tile GEMM + channel contraction, L fixed at compile time so the
// mL accumulator stays in fixed VGPRs while the c-loop stays rolled.
template <int L>
__device__ __forceinline__ void layer3_contract(const bf16_t* __restrict__ pw3,
                                                const float* __restrict__ bf3,
                                                const float* __restrict__ gbase,
                                                int lane, int hi, int m,
                                                const v16bf& a0, const v16bf& a1,
                                                v8f& mL) {
#pragma unroll 1
    for (int c = 0; c < 8; c++) {
        int t = L * 8 + c;
        v16bf b0 = *(const v16bf*)(pw3 + t * 512 + lane * 16);
        v16bf b1 = *(const v16bf*)(pw3 + (24 + t) * 512 + lane * 16);
        v8f acc = {0.f, 0.f, 0.f, 0.f, 0.f, 0.f, 0.f, 0.f};
        acc = __builtin_amdgcn_wmma_f32_16x16x32_bf16(false, a0, false, b0,
                                                      (short)0, acc, false, false);
        acc = __builtin_amdgcn_wmma_f32_16x16x32_bf16(false, a1, false, b1,
                                                      (short)0, acc, false, false);
        const float* gp = gbase + c * 16 + 8 * hi;
        v4f g0 = *(const v4f*)(gp);
        v4f g1 = *(const v4f*)(gp + 4);
        float bias = bf3[t * 16 + m];
        float gv[8] = {g0[0], g0[1], g0[2], g0[3], g1[0], g1[1], g1[2], g1[3]};
#pragma unroll
        for (int r = 0; r < 8; r++) mL[r] += gv[r] * (acc[r] + bias);
    }
}

// ---------------- main edge kernel: 16 edges per wave, WMMA MLP chain ------
__global__ __launch_bounds__(256)
void edge_kernel(const float* __restrict__ pos,
                 const float* __restrict__ shifts,
                 const float* __restrict__ cell,
                 const int* __restrict__ batch,
                 const int* __restrict__ esrc,
                 const int* __restrict__ edst,
                 const float* __restrict__ Ai,
                 const bf16_t* __restrict__ pw1,
                 const bf16_t* __restrict__ pw2,
                 const bf16_t* __restrict__ pw3,
                 const float* __restrict__ bf1,
                 const float* __restrict__ bf2,
                 const float* __restrict__ bf3,
                 float* __restrict__ out,
                 float* __restrict__ cnt,
                 int E, int ntiles) {
    __shared__ __align__(16) bf16_t sh_h[WPB][16 * 64];  // h1, then reused for h2
    __shared__ __align__(16) float  sh_g[WPB][8][16];    // gT[c][e] = Ai_src*Ai_dst
    __shared__ __align__(16) float  sh_v[WPB][16][4];    // unit edge vector
    __shared__ int                  sh_dst[WPB][16];

    const int lane = threadIdx.x & 31;
    const int wv   = threadIdx.x >> 5;
    const int hi   = lane >> 4;
    const int m    = lane & 15;

    const int tile = blockIdx.x * WPB + wv;   // one 16-edge tile per wave
    if (tile >= ntiles) return;               // whole-wave uniform exit

    float len = 0.0f;
    // ---- phase A: geometry + Ai gather (lanes 0..15, one edge each) ----
    if (hi == 0) {
        int eg = tile * 16 + m;
        float vx = 0.f, vy = 0.f, vz = 0.f;
        int d = 0;
        if (eg < E) {
            int s = esrc[eg];
            d = edst[eg];
            int b = batch[s];
            const float* cl = cell + b * 9;
            float s0 = shifts[eg * 3 + 0], s1 = shifts[eg * 3 + 1], s2 = shifts[eg * 3 + 2];
            float shx = s0 * cl[0] + s1 * cl[3] + s2 * cl[6];
            float shy = s0 * cl[1] + s1 * cl[4] + s2 * cl[7];
            float shz = s0 * cl[2] + s1 * cl[5] + s2 * cl[8];
            float ex = pos[d * 3 + 0] - pos[s * 3 + 0] + shx;
            float ey = pos[d * 3 + 1] - pos[s * 3 + 1] + shy;
            float ez = pos[d * 3 + 2] - pos[s * 3 + 2] + shz;
            len = sqrtf(ex * ex + ey * ey + ez * ez);
            float inv = __builtin_amdgcn_rcpf(len + 1e-12f);
            vx = ex * inv; vy = ey * inv; vz = ez * inv;
            const float* as = Ai + (size_t)s * 8;
            const float* ad = Ai + (size_t)d * 8;
#pragma unroll
            for (int c = 0; c < 8; c++) sh_g[wv][c][m] = as[c] * ad[c];
            atomicAdd(&cnt[d], 1.0f);
        } else {
#pragma unroll
            for (int c = 0; c < 8; c++) sh_g[wv][c][m] = 0.0f;
        }
        sh_v[wv][m][0] = vx; sh_v[wv][m][1] = vy; sh_v[wv][m][2] = vz;
        sh_dst[wv][m] = d;
    }
    __builtin_amdgcn_wave_barrier();

    // ---- layer 1: RBF(8, padded to K=32) @ Wf1 -> 16x64, silu --------
    v16bf a1;
#pragma unroll
    for (int i = 0; i < 16; i++) a1[i] = (bf16_t)0.0f;
    if (hi == 0) {
        const float stepv = 5.0f / 9.0f;
#pragma unroll
        for (int k = 0; k < 8; k++) {
            float diff = (len - (float)(k + 1) * stepv) * (9.0f / 5.0f);
            // exp(-d^2)/1.12 * sqrt(8)
            a1[k] = (bf16_t)(__expf(-diff * diff) * 2.52538136f);
        }
    }
#pragma unroll 1
    for (int n = 0; n < 4; n++) {
        v16bf b = *(const v16bf*)(pw1 + n * 512 + lane * 16);
        v8f acc = {0.f, 0.f, 0.f, 0.f, 0.f, 0.f, 0.f, 0.f};
        acc = __builtin_amdgcn_wmma_f32_16x16x32_bf16(false, a1, false, b,
                                                      (short)0, acc, false, false);
        float bias = bf1[n * 16 + m];
#pragma unroll
        for (int r = 0; r < 8; r++) {
            float h = silu_f(acc[r] + bias);
            sh_h[wv][(r + 8 * hi) * 64 + n * 16 + m] = (bf16_t)h;
        }
    }
    __builtin_amdgcn_wave_barrier();

    // ---- layer 2: 16x64 @ Wf2(64x64) -> 16x64, silu ------------------
    v16bf a2k0, a2k1;
    {
        const bf16_t* hrow = &sh_h[wv][m * 64];
        v8bf l0 = *(const v8bf*)(hrow + 8 * hi);
        v8bf l1 = *(const v8bf*)(hrow + 16 + 8 * hi);
        v8bf h0 = *(const v8bf*)(hrow + 32 + 8 * hi);
        v8bf h1v = *(const v8bf*)(hrow + 48 + 8 * hi);
#pragma unroll
        for (int i = 0; i < 8; i++) {
            a2k0[i] = l0[i]; a2k0[i + 8] = l1[i];
            a2k1[i] = h0[i]; a2k1[i + 8] = h1v[i];
        }
    }
    __builtin_amdgcn_wave_barrier();   // h1 fully consumed; buffer reused for h2
#pragma unroll 1
    for (int n = 0; n < 4; n++) {
        v16bf b0 = *(const v16bf*)(pw2 + n * 512 + lane * 16);
        v16bf b1 = *(const v16bf*)(pw2 + (4 + n) * 512 + lane * 16);
        v8f acc = {0.f, 0.f, 0.f, 0.f, 0.f, 0.f, 0.f, 0.f};
        acc = __builtin_amdgcn_wmma_f32_16x16x32_bf16(false, a2k0, false, b0,
                                                      (short)0, acc, false, false);
        acc = __builtin_amdgcn_wmma_f32_16x16x32_bf16(false, a2k1, false, b1,
                                                      (short)0, acc, false, false);
        float bias = bf2[n * 16 + m];
#pragma unroll
        for (int r = 0; r < 8; r++) {
            float h = silu_f(acc[r] + bias);
            sh_h[wv][(r + 8 * hi) * 64 + n * 16 + m] = (bf16_t)h;
        }
    }
    __builtin_amdgcn_wave_barrier();

    // ---- layer 3 + contraction: w tile (16x16) per (L,c); fold g ----
    v16bf a3k0, a3k1;
    {
        const bf16_t* hrow = &sh_h[wv][m * 64];
        v8bf l0 = *(const v8bf*)(hrow + 8 * hi);
        v8bf l1 = *(const v8bf*)(hrow + 16 + 8 * hi);
        v8bf h0 = *(const v8bf*)(hrow + 32 + 8 * hi);
        v8bf h1v = *(const v8bf*)(hrow + 48 + 8 * hi);
#pragma unroll
        for (int i = 0; i < 8; i++) {
            a3k0[i] = l0[i]; a3k0[i + 8] = l1[i];
            a3k1[i] = h0[i]; a3k1[i + 8] = h1v[i];
        }
    }
    v8f mL0 = {0.f, 0.f, 0.f, 0.f, 0.f, 0.f, 0.f, 0.f};
    v8f mL1 = {0.f, 0.f, 0.f, 0.f, 0.f, 0.f, 0.f, 0.f};
    v8f mL2 = {0.f, 0.f, 0.f, 0.f, 0.f, 0.f, 0.f, 0.f};
    const float* gbase = &sh_g[wv][0][0];
    layer3_contract<0>(pw3, bf3, gbase, lane, hi, m, a3k0, a3k1, mL0);
    layer3_contract<1>(pw3, bf3, gbase, lane, hi, m, a3k0, a3k1, mL1);
    layer3_contract<2>(pw3, bf3, gbase, lane, hi, m, a3k0, a3k1, mL2);

    // ---- scatter: out[dst] += m_L[e,o] * basis_L[e,d] ----------------
#pragma unroll
    for (int r = 0; r < 8; r++) {
        int e = r + 8 * hi;
        int eg = tile * 16 + e;
        if (eg < E) {
            int d = sh_dst[wv][e];
            float vx = sh_v[wv][e][0], vy = sh_v[wv][e][1], vz = sh_v[wv][e][2];
            float* op = out + (size_t)d * OUTW;
            int o = m;
            atomicAdd(op + o, mL0[r]);                       // L=0, cols [0,16)
            float s1 = mL1[r];
            atomicAdd(op + 16 + o * 3 + 0, s1 * vx);         // L=1, cols [16,64)
            atomicAdd(op + 16 + o * 3 + 1, s1 * vy);
            atomicAdd(op + 16 + o * 3 + 2, s1 * vz);
            float s2 = mL2[r];
            float vv[3] = {vx, vy, vz};
#pragma unroll
            for (int i2 = 0; i2 < 3; i2++)
#pragma unroll
                for (int j2 = 0; j2 < 3; j2++)               // L=2, cols [64,208)
                    atomicAdd(op + 64 + o * 9 + i2 * 3 + j2, s2 * vv[i2] * vv[j2]);
        }
    }
}

// ---------------- normalize by clipped in-degree ---------------------------
__global__ void norm_kernel(float* __restrict__ out, const float* __restrict__ cnt,
                            int total) {
    int i = blockIdx.x * blockDim.x + threadIdx.x;
    if (i >= total) return;
    int n = i / OUTW;
    out[i] = out[i] * __builtin_amdgcn_rcpf(fmaxf(cnt[n], 1.0f));
}

extern "C" void kernel_launch(void* const* d_in, const int* in_sizes, int n_in,
                              void* d_out, int out_size, void* d_ws, size_t ws_size,
                              hipStream_t stream) {
    const float* pos       = (const float*)d_in[0];
    const float* shifts    = (const float*)d_in[1];
    const float* cell      = (const float*)d_in[2];
    const int*   A         = (const int*)d_in[3];
    const int*   batch     = (const int*)d_in[4];
    const int*   esrc      = (const int*)d_in[5];
    const int*   edst      = (const int*)d_in[6];
    const float* emb_table = (const float*)d_in[7];
    const float* Wa1       = (const float*)d_in[8];
    const float* ba1       = (const float*)d_in[9];
    const float* Wa2       = (const float*)d_in[10];
    const float* ba2       = (const float*)d_in[11];
    const float* Wf1       = (const float*)d_in[12];
    const float* bf1       = (const float*)d_in[13];
    const float* Wf2       = (const float*)d_in[14];
    const float* bf2       = (const float*)d_in[15];
    const float* Wf3       = (const float*)d_in[16];
    const float* bf3       = (const float*)d_in[17];

    const int N = in_sizes[3];
    const int E = in_sizes[5];
    float* outp = (float*)d_out;

    // workspace layout: Ai[N*8] | cnt[N] | packed bf16 weights (32B aligned)
    float* Ai  = (float*)d_ws;
    float* cnt = Ai + (size_t)N * 8;
    uintptr_t p = (uintptr_t)(cnt + N);
    p = (p + 31) & ~(uintptr_t)31;
    bf16_t* pw1 = (bf16_t*)p;
    bf16_t* pw2 = pw1 + 4 * 512;
    bf16_t* pw3 = pw2 + 8 * 512;

    zero_kernel<<<(out_size + 255) / 256, 256, 0, stream>>>(outp, out_size, cnt, N);
    node_mlp_kernel<<<(N + 255) / 256, 256, 0, stream>>>(A, emb_table, Wa1, ba1,
                                                         Wa2, ba2, Ai, N);
    const int npack = (4 + 8 + 48) * 512;
    pack_weights_kernel<<<(npack + 255) / 256, 256, 0, stream>>>(Wf1, Wf2, Wf3,
                                                                 pw1, pw2, pw3);
    const int ntiles = (E + 15) / 16;
    const int blocks = (ntiles + WPB - 1) / WPB;
    edge_kernel<<<blocks, 256, 0, stream>>>(pos, shifts, cell, batch, esrc, edst,
                                            Ai, pw1, pw2, pw3, bf1, bf2, bf3,
                                            outp, cnt, E, ntiles);
    norm_kernel<<<(out_size + 255) / 256, 256, 0, stream>>>(outp, cnt, out_size);
}
